// Relations_13022340841937
// MI455X (gfx1250) — matrix operand order repaired and checked by reading, not compiled, not run on hardware
//
#include <hip/hip_runtime.h>

#define GG 20
#define LL 32
#define HH 30
#define CC 190
#define BB 8192
#define FF 2

// Prepacked weight image size (also the first 22784 bytes of LDS)
#define WIMG_BYTES 22784

typedef __fp16 __attribute__((ext_vector_type(2)))    v2hf;  // matches cvt_pkrtz return type
typedef __attribute__((ext_vector_type(4)))  _Float16 v4h;
typedef __attribute__((ext_vector_type(8)))  _Float16 v8h;
typedef __attribute__((ext_vector_type(16))) _Float16 v16h;
typedef __attribute__((ext_vector_type(8)))  float    v8f;

static __device__ __forceinline__ v16h cat8(v8h a, v8h b) {
  return __builtin_shufflevector(a, b, 0,1,2,3,4,5,6,7,8,9,10,11,12,13,14,15);
}
static __device__ __forceinline__ v8f wmma_f16(v16h a, v16h b, v8f c) {
  // D = A(16x32 f16) * B(32x16 f16) + C(16x16 f32)
  return __builtin_amdgcn_wmma_f32_16x16x32_f16(false, a, false, b, (short)0, c, false, false);
}
// Four independent v_tanh_f32: each covers the next one's 1-instr trans hazard;
// trailing v_nop covers the last. Early-clobber so outputs don't alias inputs.
static __device__ __forceinline__ void tanh4_hw(float a, float b, float c, float d,
                                                float& ra, float& rb, float& rc, float& rd) {
  asm volatile("v_tanh_f32 %0, %4\n\t"
               "v_tanh_f32 %1, %5\n\t"
               "v_tanh_f32 %2, %6\n\t"
               "v_tanh_f32 %3, %7\n\t"
               "v_nop"
               : "=&v"(ra), "=&v"(rb), "=&v"(rc), "=&v"(rd)
               : "v"(a), "v"(b), "v"(c), "v"(d));
}

// Hidden-neuron storage permutation: storage column p = 2*N + t holds original
// neuron (16*t + N). Act columns and W1..W4 K-rows use storage order; weight
// columns (and b0) stay in natural order. WMMA K-sums are permutation-invariant.
// orig(p) = 16*(p&1) + (p>>1).  Bias slot orig-30 -> p=29; pad orig-31 -> p=31.

// Weight image layout (bytes) — identical in d_ws and in LDS[0..22784):
//  [0,22528)      padded f16 weights, per f (11264 B each):
//     W0p @ +0     : 64 rows x 32 cols (row stride 64B)        [rows natural]
//     W1p @ +4096, W2p @ +6144, W3p @ +8192 : 32x32            [rows permuted]
//     W4p @ +10240 : 32 rows x 16 cols (col0 = W4/b4, rest 0)  [rows permuted]
//  [22528,22784)  b0 padded f32: [F][32] (natural order)
// LDS additionally:
//  [22784,41216)  per-wave pair buffer: 8 x (16 rows x 144B stride)
//  [41216,51456)  per-wave act buffer:  8 x (16 rows x 80B stride)
#define OFF_B0   22528
#define OFF_PAIR 22784
#define OFF_ACT  41216

// ---------------- prep kernel: build padded f16 weight image once ----------------
__global__ __launch_bounds__(256) void relations_prep_kernel(
    const float* __restrict__ W0, const float* __restrict__ b0,
    const float* __restrict__ W1, const float* __restrict__ b1,
    const float* __restrict__ W2, const float* __restrict__ b2,
    const float* __restrict__ W3, const float* __restrict__ b3,
    const float* __restrict__ W4, const float* __restrict__ b4,
    unsigned char* __restrict__ wimg)
{
  const int tid = threadIdx.x;
  for (int idx = tid; idx < FF*64*32; idx += 256) {            // W0: [F,64,30] -> 64x32
    int f = idx >> 11, r = (idx >> 5) & 63, n = idx & 31;
    float v = (n < HH) ? W0[(f*64 + r)*HH + n] : 0.f;
    *(_Float16*)(wimg + f*11264 + ((r >> 5)*32 + ((r >> 3) & 1)*16 + (n & 15))*64 + (n >> 4)*32 + ((r & 7) + ((r >> 4) & 1)*8)*2) = (_Float16)v;
  }
  {
    const float* Wg[3] = {W1, W2, W3};
    const float* bg[3] = {b1, b2, b3};
    for (int l = 0; l < 3; ++l)
      for (int idx = tid; idx < FF*32*32; idx += 256) {        // W1..3: K-rows permuted
        int f = idx >> 10, p = (idx >> 5) & 31, n = idx & 31;
        int k = ((p & 1) << 4) | (p >> 1);                     // orig neuron for row p
        float v = 0.f;
        if (n < HH) {
          if (k < HH)       v = Wg[l][(f*HH + k)*HH + n];
          else if (k == HH) v = bg[l][f*HH + n];
        }
        *(_Float16*)(wimg + f*11264 + 4096 + l*2048 + (((p >> 3) & 1)*16 + (n & 15))*64 + (n >> 4)*32 + ((p & 7) + ((p >> 4) & 1)*8)*2) = (_Float16)v;
      }
  }
  for (int idx = tid; idx < FF*32*16; idx += 256) {            // W4: 32x16, col 0 only
    int f = idx >> 9, p = (idx >> 4) & 31, n = idx & 15;
    int k = ((p & 1) << 4) | (p >> 1);
    float v = 0.f;
    if (n == 0) { if (k < HH) v = W4[f*HH + k]; else if (k == HH) v = b4[f]; }
    *(_Float16*)(wimg + f*11264 + 10240 + (((p >> 3) & 1)*16 + n)*32 + ((p & 7) + ((p >> 4) & 1)*8)*2) = (_Float16)v;
  }
  for (int idx = tid; idx < FF*32; idx += 256) {               // b0 padded f32 (natural)
    int f = idx >> 5, n = idx & 31;
    *(float*)(wimg + OFF_B0 + idx*4) = (n < HH) ? b0[f*HH + n] : 0.f;
  }
}

// ---------------- main kernel ----------------
__global__ __launch_bounds__(256) void relations_wmma_kernel(
    const float* __restrict__ x,
    const unsigned char* __restrict__ wimg,
    float* __restrict__ out)
{
  __shared__ __align__(16) unsigned char lds[51456];
  const int tid = threadIdx.x;

  // copy prepacked weight image global -> LDS (1424 x 16B)
  {
    const uint4* src = (const uint4*)wimg;
    for (int i = tid; i < WIMG_BYTES/16; i += 256)
      *(uint4*)(lds + i*16) = src[i];
  }
  __syncthreads();

  // ---------- per-wave tile: 16 batch rows x one pair c, loop over f ----------
  const int wid  = tid >> 5;
  const int lane = tid & 31;
  const int mrow = lane & 15;      // D-tile N index / A row id
  const int hsel = lane >> 4;      // A-layout half select (K +8 for lanes 16..31)
  const unsigned u = blockIdx.x * 8u + wid;   // unit id in [0, 97280)
  const int bt    = u & 511;
  const int cpair = u >> 9;
  const int brow0 = bt << 4;
  int ci = 0, rem = cpair;                    // triu_indices(G,1) order
  while (rem >= (GG - 1 - ci)) { rem -= (GG - 1 - ci); ++ci; }
  const int cj = ci + 1 + rem;

  unsigned char* pb = lds + OFF_PAIR + wid*2304;  // 16 x 64 f16, stride 144B
  unsigned char* ab = lds + OFF_ACT  + wid*1280;  // 16 x 32 f16, stride 80B
  unsigned char* sb = ab + hsel*8*80 + mrow*4;    // this lane's act store base

  // stage pairs: rows m: [ xg[b,i,0..31] | xg[b,j,0..31] ] as f16 row-major
  {
    const float* xr = x + (size_t)brow0 * (GG*LL);
    for (int t = lane; t < 256; t += 32) {
      int m = t >> 4, q = t & 15;
      int col = (q < 8) ? (ci*LL + q*4) : (cj*LL + (q-8)*4);
      const float4 v = *(const float4*)(xr + (size_t)m*(GG*LL) + col);
      v4h h = { (_Float16)v.x, (_Float16)v.y, (_Float16)v.z, (_Float16)v.w };
      *(v4h*)(pb + m*144 + q*8) = h;
    }
  }
  asm volatile("" ::: "memory");   // in-wave LDS is in-order; stop compiler reordering

  for (int f = 0; f < FF; ++f) {
    const unsigned char* wb = lds + f*11264;

    // ---- layer 0: [16x64] x [64x32], K split into two 32-chunks ----
    const unsigned char* pr = pb + mrow*144 + hsel*16;
    v16h A0 = cat8(*(const v8h*)(pr +  0), *(const v8h*)(pr + 32));  // K 0..31
    v16h A1 = cat8(*(const v8h*)(pr + 64), *(const v8h*)(pr + 96));  // K 32..63
    v16h Bk0t0 = *(const v16h*)(wb + lane*64);
    v16h Bk0t1 = *(const v16h*)(wb + lane*64 + 32);
    v16h Bk1t0 = *(const v16h*)(wb + (32+lane)*64);
    v16h Bk1t1 = *(const v16h*)(wb + (32+lane)*64 + 32);
    v8f acc0 = {}, acc1 = {};
    acc0 = wmma_f16(A0, Bk0t0, acc0);
    acc0 = wmma_f16(A1, Bk1t0, acc0);
    acc1 = wmma_f16(A0, Bk0t1, acc1);
    acc1 = wmma_f16(A1, Bk1t1, acc1);
    float bi0 = *(const float*)(lds + OFF_B0 + (f*32 + mrow)*4);
    float bi1 = *(const float*)(lds + OFF_B0 + (f*32 + 16 + mrow)*4);
#pragma unroll
    for (int r = 0; r < 8; r += 2) {
      float t0, t1, t2, t3;
      tanh4_hw(acc0[r] + bi0, acc1[r] + bi1, acc0[r+1] + bi0, acc1[r+1] + bi1,
               t0, t1, t2, t3);
      v2hf pA = __builtin_amdgcn_cvt_pkrtz(t0, t1);  // .l -> p=2*mrow, .h -> p=2*mrow+1
      v2hf pB = __builtin_amdgcn_cvt_pkrtz(t2, t3);
      *(v2hf*)(sb + r*80)      = pA;                 // adjacent stores, shared base:
      *(v2hf*)(sb + (r+1)*80)  = pB;                 //  mergeable to ds_store_2addr_b32
    }
    if (lane < 16) {                                 // storage p=29 (orig 30): act=1; p=31: 0
      *(_Float16*)(ab + lane*80 + 58) = (_Float16)1.0f;
      *(_Float16*)(ab + lane*80 + 62) = (_Float16)0.0f;
    }
    asm volatile("" ::: "memory");

    // ---- layers 1..3: [16x32] x [32x32], bias folded in (permuted) row of orig-30 ----
#pragma unroll
    for (int l = 0; l < 3; ++l) {
      const unsigned char* ar = ab + mrow*80 + hsel*16;
      v16h A = cat8(*(const v8h*)(ar + 0), *(const v8h*)(ar + 32));
      const unsigned char* wl = wb + 4096 + l*2048;
      v16h Bt0 = *(const v16h*)(wl + lane*64);
      v16h Bt1 = *(const v16h*)(wl + lane*64 + 32);
      v8f z = {};
      v8f n0 = wmma_f16(A, Bt0, z);
      v8f n1 = wmma_f16(A, Bt1, z);
#pragma unroll
      for (int r = 0; r < 8; r += 2) {
        float t0, t1, t2, t3;
        tanh4_hw(n0[r], n1[r], n0[r+1], n1[r+1], t0, t1, t2, t3);
        v2hf pA = __builtin_amdgcn_cvt_pkrtz(t0, t1);
        v2hf pB = __builtin_amdgcn_cvt_pkrtz(t2, t3);
        *(v2hf*)(sb + r*80)     = pA;
        *(v2hf*)(sb + (r+1)*80) = pB;
      }
      if (lane < 16) {
        *(_Float16*)(ab + lane*80 + 58) = (_Float16)1.0f;
        *(_Float16*)(ab + lane*80 + 62) = (_Float16)0.0f;
      }
      asm volatile("" ::: "memory");
    }

    // ---- layer 4: [16x32] x [32x16] (col0 = W4+b4, K permuted), sigmoid, store ----
    {
      const unsigned char* ar = ab + mrow*80 + hsel*16;
      v16h A = cat8(*(const v8h*)(ar + 0), *(const v8h*)(ar + 32));
      v16h B4v = *(const v16h*)(wb + 10240 + lane*32);
      v8f z = {};
      v8f n4 = wmma_f16(A, B4v, z);
      if (mrow == 0) {                       // N==0 lives in lanes 0 and 16
#pragma unroll
        for (int r = 0; r < 8; ++r) {
          int m = r + hsel*8;
          float s = 1.0f / (1.0f + __expf(-n4[r]));
          // output flat: b*(C*F) + f*C + c  (matches reference reshape order)
          out[(size_t)(brow0 + m)*(CC*FF) + f*CC + cpair] = s;
        }
      }
    }
    asm volatile("" ::: "memory");
  }
}

extern "C" void kernel_launch(void* const* d_in, const int* in_sizes, int n_in,
                              void* d_out, int out_size, void* d_ws, size_t ws_size,
                              hipStream_t stream) {
  const float* x  = (const float*)d_in[0];
  const float* W0 = (const float*)d_in[1];
  const float* b0 = (const float*)d_in[2];
  const float* W1 = (const float*)d_in[3];
  const float* b1 = (const float*)d_in[4];
  const float* W2 = (const float*)d_in[5];
  const float* b2 = (const float*)d_in[6];
  const float* W3 = (const float*)d_in[7];
  const float* b3 = (const float*)d_in[8];
  const float* W4 = (const float*)d_in[9];
  const float* b4 = (const float*)d_in[10];
  unsigned char* wimg = (unsigned char*)d_ws;     // needs WIMG_BYTES (22784 B) of scratch

  relations_prep_kernel<<<dim3(1), dim3(256), 0, stream>>>(
      W0, b0, W1, b1, W2, b2, W3, b3, W4, b4, wimg);

  // units = (B/16)*C = 512*190 = 97280 waves; 8 waves per 256-thread block
  dim3 grid(((BB/16) * CC) / 8);   // 12160 blocks, exact cover, no bounds checks
  relations_wmma_kernel<<<grid, dim3(256), 0, stream>>>(x, wimg, (float*)d_out);
}
